// LokrLinear_81750407512599
// MI455X (gfx1250) — matrix operand order repaired and checked by reading, not compiled, not run on hardware
//
#include <hip/hip_runtime.h>

typedef __attribute__((ext_vector_type(16))) __bf16 v16bf;
typedef __attribute__((ext_vector_type(8)))  __bf16 v8bf;
typedef __attribute__((ext_vector_type(8)))  float  v8f;
typedef __attribute__((ext_vector_type(4)))  float  f4;

#define K_DIM   4096
#define N_DIM   4096
#define BM      128
#define BN      128
#define BK      32
#define LDK     40          // BK + 8 bf16 pad -> 80B row stride (16B aligned, conflict-spread)
#define SCALING 2.0f        // ALPHA / R = 32 / 16

static __device__ __forceinline__ unsigned pk2(__bf16 a, __bf16 b) {
  return (unsigned)__builtin_bit_cast(unsigned short, a) |
         ((unsigned)__builtin_bit_cast(unsigned short, b) << 16);
}

static __device__ __forceinline__ v16bf frag16(const __bf16* p0, const __bf16* p1) {
  v8bf a = *(const v8bf*)p0;
  v8bf b = *(const v8bf*)p1;
  return __builtin_shufflevector(a, b, 0,1,2,3,4,5,6,7,8,9,10,11,12,13,14,15);
}

// wl = SCALING * (lora_A @ lora_B), 16x16 — one tiny block
__global__ __launch_bounds__(256) void lokr_wleft(const float* __restrict__ A,
                                                  const float* __restrict__ B,
                                                  float* __restrict__ wl) {
  int t = threadIdx.x;
  int i = t >> 4, j = t & 15;
  float s = 0.f;
#pragma unroll
  for (int k = 0; k < 16; ++k) s += A[i * 16 + k] * B[k * 16 + j];
  wl[t] = s * SCALING;
}

// out[M,N] = x[M,K] @ (W + wl[k>>8,n>>8]*O[k&255,n&255]) + bias
// bf16x3 split-precision GEMM on v_wmma_f32_16x16x32_bf16.
__global__ __launch_bounds__(256) void lokr_gemm(const float* __restrict__ x,
                                                 const float* __restrict__ w,
                                                 const float* __restrict__ bias,
                                                 const float* __restrict__ O,
                                                 const float* __restrict__ wl,
                                                 float* __restrict__ out) {
  __shared__ __bf16 Ah[BM * LDK];
  __shared__ __bf16 Al[BM * LDK];
  __shared__ __bf16 Bh[BN * LDK];   // transposed: [n][k] so B fragments are K-contiguous
  __shared__ __bf16 Bl[BN * LDK];

  const int tid   = threadIdx.x;
  const int lane  = tid & 31;
  const int wv    = tid >> 5;              // 8 waves
  const int n_blk = blockIdx.x * BN;       // fast dim: x-panel reuse in L2
  const int m_blk = blockIdx.y * BM;

  // global-load coordinates
  const int a_row = tid >> 3;              // 0..31, x4 passes
  const int a_kq  = (tid & 7) << 2;        // k-quad within BK
  const int b_nq  = (tid & 31) << 2;       // n-quad within BN
  const int b_k2  = (tid >> 5) << 1;       // even k within half-tile

  // wave tiling: 4(M) x 2(N) waves; each wave 32x64 = 2x4 tiles of 16x16
  const int mwb  = (wv >> 1) * 32;
  const int nwb  = (wv & 1) * 64;
  const int lrow = lane & 15;
  const int kp   = lane >> 4;              // lane half selects K parity per ISA layout

  v8f acc[2][4] = {};

  const int wl_col = n_blk >> 8;
  const int o_ncol = n_blk & 255;

  for (int kt = 0; kt < K_DIM / BK; ++kt) {
    const int k0 = kt * BK;
    const float wls = wl[(kt >> 3) * 16 + wl_col];   // uniform per iteration -> s_load

    // ---- stage global loads in registers (coalesced float4) ----
    f4 ar[4];
#pragma unroll
    for (int p = 0; p < 4; ++p) {
      const int row = a_row + p * 32;
      ar[p] = *(const f4*)(x + (size_t)(m_blk + row) * K_DIM + k0 + a_kq);
    }
    f4 br[2][2], orr[2][2];
#pragma unroll
    for (int p = 0; p < 2; ++p) {
      const int k = p * 16 + b_k2;
      const float* bp = w + (size_t)(k0 + k) * N_DIM + n_blk + b_nq;
      br[p][0] = *(const f4*)bp;
      br[p][1] = *(const f4*)(bp + N_DIM);
      const float* op = O + (size_t)((k0 + k) & 255) * 256 + o_ncol + b_nq;
      orr[p][0] = *(const f4*)op;
      orr[p][1] = *(const f4*)(op + 256);
    }

    __syncthreads();   // previous iteration's fragment reads done

    // ---- fp32 -> (hi,lo) bf16 split, write LDS ----
#pragma unroll
    for (int p = 0; p < 4; ++p) {
      const int row = a_row + p * 32;
      __bf16 h[4], l[4];
#pragma unroll
      for (int j = 0; j < 4; ++j) {
        float v = ar[p][j];
        h[j] = (__bf16)v;
        l[j] = (__bf16)(v - (float)h[j]);
      }
      uint2 hp = { pk2(h[0], h[1]), pk2(h[2], h[3]) };
      uint2 lp = { pk2(l[0], l[1]), pk2(l[2], l[3]) };
      *(uint2*)&Ah[row * LDK + a_kq] = hp;
      *(uint2*)&Al[row * LDK + a_kq] = lp;
    }
#pragma unroll
    for (int p = 0; p < 2; ++p) {
      const int k = p * 16 + b_k2;
#pragma unroll
      for (int j = 0; j < 4; ++j) {
        // fused LoKr: W_eff = W + wls * O  (never materialized in HBM)
        float w0 = br[p][0][j] + wls * orr[p][0][j];
        float w1 = br[p][1][j] + wls * orr[p][1][j];
        __bf16 h0 = (__bf16)w0, h1 = (__bf16)w1;
        __bf16 l0 = (__bf16)(w0 - (float)h0), l1 = (__bf16)(w1 - (float)h1);
        *(unsigned*)&Bh[(b_nq + j) * LDK + k] = pk2(h0, h1);
        *(unsigned*)&Bl[(b_nq + j) * LDK + k] = pk2(l0, l1);
      }
    }

    __syncthreads();

    // ---- fragments per ISA 16-bit A/B VGPR layouts ----
    v16bf ah[2], al2[2], bh[4], bl[4];
#pragma unroll
    for (int i = 0; i < 2; ++i) {
      const int r = (mwb + i * 16 + lrow) * LDK;
      ah[i]  = frag16(&Ah[r + kp * 8], &Ah[r + 16 + kp * 8]);
      al2[i] = frag16(&Al[r + kp * 8], &Al[r + 16 + kp * 8]);
    }
#pragma unroll
    for (int j = 0; j < 4; ++j) {
      const int c = (nwb + j * 16 + lrow) * LDK + kp * 8;
      bh[j] = frag16(&Bh[c], &Bh[c + 16]);
      bl[j] = frag16(&Bl[c], &Bl[c + 16]);
    }

    // ---- bf16x3: A*B ≈ Al*Bh + Ah*Bl + Ah*Bh, fp32 accumulate ----
#pragma unroll
    for (int i = 0; i < 2; ++i)
#pragma unroll
      for (int j = 0; j < 4; ++j) {
        acc[i][j] = __builtin_amdgcn_wmma_f32_16x16x32_bf16(
            false, al2[i], false, bh[j], (short)0, acc[i][j], false, false);
        acc[i][j] = __builtin_amdgcn_wmma_f32_16x16x32_bf16(
            false, ah[i], false, bl[j], (short)0, acc[i][j], false, false);
        acc[i][j] = __builtin_amdgcn_wmma_f32_16x16x32_bf16(
            false, ah[i], false, bh[j], (short)0, acc[i][j], false, false);
      }
  }

  // ---- epilogue: bias add + store (C/D layout: lane half picks M+8) ----
#pragma unroll
  for (int j = 0; j < 4; ++j) {
    const int col = n_blk + nwb + j * 16 + lrow;
    const float bv = bias[col];
#pragma unroll
    for (int i = 0; i < 2; ++i) {
      const int rbase = m_blk + mwb + i * 16 + kp * 8;
#pragma unroll
      for (int r = 0; r < 8; ++r)
        out[(size_t)(rbase + r) * N_DIM + col] = acc[i][j][r] + bv;
    }
  }
}

extern "C" void kernel_launch(void* const* d_in, const int* in_sizes, int n_in,
                              void* d_out, int out_size, void* d_ws, size_t ws_size,
                              hipStream_t stream) {
  (void)in_sizes; (void)n_in; (void)out_size; (void)ws_size;
  const float* x      = (const float*)d_in[0];
  const float* w      = (const float*)d_in[1];
  const float* bias   = (const float*)d_in[2];
  const float* lora_A = (const float*)d_in[3];
  const float* lora_B = (const float*)d_in[4];
  const float* O      = (const float*)d_in[5];
  float* out = (float*)d_out;
  float* wl  = (float*)d_ws;          // 256 floats

  lokr_wleft<<<1, 256, 0, stream>>>(lora_A, lora_B, wl);

  dim3 grid(N_DIM / BN, (4 * 4096) / BM);  // (32, 128); n fast for L2 panel reuse
  lokr_gemm<<<grid, 256, 0, stream>>>(x, w, bias, O, wl, out);
}
